// DenseGraphTransformerModel_23450521436194
// MI455X (gfx1250) — matrix operand order repaired and checked by reading, not compiled, not run on hardware
//
#include <hip/hip_runtime.h>

#define N_ROWS 8192
#define IN_DIM 128
#define HID    128
#define HEADS  4
#define HD     32
#define OUT_DIM 40

#define QBLK 64                    // query rows per block (4 waves x 16 per head)
#define SEG  2                     // key-dimension split (flash-decoding)
#define KSEG (N_ROWS / SEG)        // keys per segment
#define KPAD 40                    // LDS row pitch in halves (80B: 16B aligned)

typedef __attribute__((ext_vector_type(16))) _Float16 v16h;
typedef __attribute__((ext_vector_type(2)))  __fp16   v2fp16;   // cvt_pkrtz result type
typedef __attribute__((ext_vector_type(8)))  float    v8f;

// bias = scale * nan_to_num(1/nan_to_num(sp, -1), 0)  (cheap v_rcp_f32 form)
__device__ __forceinline__ float bias_of(float sv, float bscale) {
    if (!(fabsf(sv) < 3.0e38f)) sv = -1.0f;            // nan/±inf -> -1
    float r = __builtin_amdgcn_rcpf(sv);               // v_rcp_f32
    if (!(fabsf(r) < 3.0e38f)) r = 0.0f;               // 1/0, nan -> 0
    return bscale * r;
}

// ---------------------------------------------------------------------------
// Kernel 1: h = x @ W_in^T + b_in ; qkv = h @ in_proj_w^T + in_proj_b
// q,k stored [H][N][HD] f16;  v stored TRANSPOSED [H][HD][N] f16.
// ---------------------------------------------------------------------------
__global__ __launch_bounds__(128) void proj_qkv_kernel(
    const float* __restrict__ x, const float* __restrict__ W_in, const float* __restrict__ b_in,
    const float* __restrict__ ipw, const float* __restrict__ ipb,
    float* __restrict__ h_out,
    _Float16* __restrict__ q, _Float16* __restrict__ k, _Float16* __restrict__ vT)
{
    __shared__ float xs[IN_DIM];
    __shared__ float hs[HID];
    const int row = blockIdx.x;
    const int t   = threadIdx.x;

    xs[t] = x[(size_t)row * IN_DIM + t];
    __syncthreads();

    float acc = b_in[t];
    #pragma unroll 8
    for (int i = 0; i < IN_DIM; ++i) acc += xs[i] * W_in[t * IN_DIM + i];
    hs[t] = acc;
    h_out[(size_t)row * HID + t] = acc;
    __syncthreads();

    const int head = t / HD, d = t % HD;
    #pragma unroll
    for (int j = 0; j < 3; ++j) {
        const int oc = j * HID + t;
        float a = ipb[oc];
        #pragma unroll 8
        for (int i = 0; i < HID; ++i) a += hs[i] * ipw[oc * HID + i];
        const _Float16 hv = (_Float16)a;
        if (j == 0)      q[((size_t)head * N_ROWS + row) * HD + d] = hv;
        else if (j == 1) k[((size_t)head * N_ROWS + row) * HD + d] = hv;
        else             vT[((size_t)head * HD + d) * N_ROWS + row] = hv;
    }
}

// ---------------------------------------------------------------------------
// Kernel 2: flash attention.  One block = 512 threads = 16 waves =
// 4 heads x 4 query tiles (64 query rows).  Per 32-key chunk the block
// cooperatively stages the sp tile (bias computed ONCE) and the K/V chunk of
// ALL 4 heads in LDS.  Score tiles are split EVEN/ODD keys so each lane's
// two exp() results are adjacent keys -> packed f16 P stores and float2 bias
// loads, while P columns / V rows stay in natural (contiguous) key order.
// Key dimension split in SEG segments (partials combined in head kernel).
// WMMA layouts per CDNA5 ISA 7.12.2 (wave32).
// ---------------------------------------------------------------------------
__global__ __launch_bounds__(512) void flash_attn_kernel(
    const float* __restrict__ sp, const float* __restrict__ bscale_p,
    const _Float16* __restrict__ q, const _Float16* __restrict__ k,
    const _Float16* __restrict__ vT,
    float* __restrict__ o_part, float* __restrict__ m_part, float* __restrict__ l_part)
{
    __shared__ __align__(16) float    lds_bias[QBLK * 32];          // 8KB
    __shared__ __align__(16) _Float16 lds_K[HEADS][32][KPAD];       // [head][key][dim]
    __shared__ __align__(16) _Float16 lds_V[HEADS][32][KPAD];       // [head][dim][key]
    __shared__ __align__(16) _Float16 lds_p[16][16 * KPAD];         // per-wave P staging

    const int tid  = threadIdx.x;
    const int w    = tid >> 5;            // wave 0..15
    const int head = w & 3;
    const int qt   = w >> 2;              // query tile 0..3
    const int lane = tid & 31;
    const int lm   = lane & 15;
    const int lg   = lane >> 4;

    const int qg  = blockIdx.x * QBLK;    // block's first query row
    const int qb  = qg + qt * 16;         // wave's first query row
    const int seg = blockIdx.y;
    const int kb0 = seg * KSEG;

    const float bscale = bscale_p[0];
    const float sscale = 0.17677669529663687f;   // 1/sqrt(HD)

    // cooperative mappings (512 threads)
    const int br  = tid >> 3;             // sp row 0..63
    const int bc  = (tid & 7) * 4;        // sp col 0,4,..,28
    const int sh  = tid >> 7;             // staging head 0..3
    const int sk  = (tid >> 2) & 31;      // staging key (K) / dim (V) 0..31
    const int spt = (tid & 3) * 8;        // staging 8-half part

    // ---- Q tile in WMMA A layout (held for whole kernel) ----
    const _Float16* qp = q + ((size_t)head * N_ROWS + qb) * HD;
    v16h aQ;
    #pragma unroll
    for (int i = 0; i < 8; ++i) {
        aQ[i]     = qp[lm * HD + 8 * lg + i];
        aQ[8 + i] = qp[lm * HD + 16 + 8 * lg + i];
    }

    float mrun = -__builtin_inff();       // shared stabilizer per lane-half
    float lacc[8];                        // per-lane partial row sums
    v8f o0 = {}, o1 = {};
    #pragma unroll
    for (int r = 0; r < 8; ++r) lacc[r] = 0.0f;

    _Float16* pl = &lds_p[w][0];

    for (int kk = 0; kk < KSEG; kk += 32) {
        const int kb = kb0 + kk;

        // ---- global loads for this chunk (issued before any barrier) ----
        const float4 s4 = *(const float4*)(sp + (size_t)(qg + br) * N_ROWS + kb + bc);
        const float4 kq = *(const float4*)(k  + ((size_t)sh * N_ROWS + kb + sk) * HD + spt);
        const float4 vq = *(const float4*)(vT + ((size_t)sh * HD + sk) * N_ROWS + kb + spt);
        if (kk + 32 < KSEG) {
            __builtin_prefetch(sp + (size_t)(qg + br) * N_ROWS + kb + 32 + bc, 0, 0);
            __builtin_prefetch(k  + ((size_t)sh * N_ROWS + kb + 32 + sk) * HD + spt, 0, 0);
            __builtin_prefetch(vT + ((size_t)sh * HD + sk) * N_ROWS + kb + 32 + spt, 0, 0);
        }
        float4 b4;
        b4.x = bias_of(s4.x, bscale);  b4.y = bias_of(s4.y, bscale);
        b4.z = bias_of(s4.z, bscale);  b4.w = bias_of(s4.w, bscale);

        // ---- publish shared tiles ----
        __syncthreads();                               // prior readers done
        *(float4*)(lds_bias + br * 32 + bc) = b4;
        *(float4*)(&lds_K[sh][sk][spt]) = kq;
        *(float4*)(&lds_V[sh][sk][spt]) = vq;
        __syncthreads();                               // tiles visible

        // ---- per-wave B tiles: score tiles take EVEN/ODD key rows ----
        v16h bK0, bK1, bV0, bV1;
        #pragma unroll
        for (int i = 0; i < 2; ++i) {
            ((float4*)&bK0)[i] = *(const float4*)(&lds_K[head][2 * lm][8 * lg + 16 * i]);
            ((float4*)&bK1)[i] = *(const float4*)(&lds_K[head][2 * lm + 1][8 * lg + 16 * i]);
            ((float4*)&bV0)[i] = *(const float4*)(&lds_V[head][lm][8 * lg + 16 * i]);
            ((float4*)&bV1)[i] = *(const float4*)(&lds_V[head][16 + lm][8 * lg + 16 * i]);
        }
        v8f s0 = {}, s1 = {};
        s0 = __builtin_amdgcn_wmma_f32_16x16x32_f16(false, aQ, false, bK0, (short)0, s0, false, false);
        s1 = __builtin_amdgcn_wmma_f32_16x16x32_f16(false, aQ, false, bK1, (short)0, s1, false, false);

        // ---- scores + bias (float2 loads), online softmax ----
        float t0[8], t1[8];
        float mt = -__builtin_inff();
        #pragma unroll
        for (int r = 0; r < 8; ++r) {
            const float2 bb = *(const float2*)(lds_bias + (qt * 16 + r + 8 * lg) * 32 + 2 * lm);
            t0[r] = s0[r] * sscale + bb.x;             // key 2*lm
            t1[r] = s1[r] * sscale + bb.y;             // key 2*lm+1
            mt = fmaxf(mt, fmaxf(t0[r], t1[r]));
        }
        #pragma unroll
        for (int off = 1; off < 16; off <<= 1) mt = fmaxf(mt, __shfl_xor(mt, off, 32));

        if (__any(mt > mrun)) {                        // rare after warm-up
            const float mnew  = fmaxf(mrun, mt);
            const float alpha = __expf(mrun - mnew);
            #pragma unroll
            for (int r = 0; r < 8; ++r) {
                o0[r] *= alpha;  o1[r] *= alpha;  lacc[r] *= alpha;
            }
            mrun = mnew;
        }

        #pragma unroll
        for (int r = 0; r < 8; ++r) {
            const float e0 = __expf(t0[r] - mrun);
            const float e1 = __expf(t1[r] - mrun);
            lacc[r] += e0 + e1;                        // reduce across lanes at end
            // adjacent keys -> one packed store, P columns in natural key order
            *(v2fp16*)(pl + (r + 8 * lg) * KPAD + 2 * lm) = __builtin_amdgcn_cvt_pkrtz(e0, e1);
        }
        asm volatile("s_wait_dscnt 0" ::: "memory");   // own-wave LDS region RAW

        // ---- P in A layout, O += P @ V ----
        v16h aP;
        ((float4*)&aP)[0] = *(const float4*)(pl + lm * KPAD + 8 * lg);
        ((float4*)&aP)[1] = *(const float4*)(pl + lm * KPAD + 16 + 8 * lg);
        o0 = __builtin_amdgcn_wmma_f32_16x16x32_f16(false, aP, false, bV0, (short)0, o0, false, false);
        o1 = __builtin_amdgcn_wmma_f32_16x16x32_f16(false, aP, false, bV1, (short)0, o1, false, false);
    }

    // ---- epilogue: reduce row sums, store unnormalized O + (m,l) partials ----
    #pragma unroll
    for (int r = 0; r < 8; ++r) {
        float ls = lacc[r];
        #pragma unroll
        for (int off = 1; off < 16; off <<= 1) ls += __shfl_xor(ls, off, 32);
        const int row = qb + r + 8 * lg;
        float* op = o_part + ((size_t)seg * N_ROWS + row) * HID + head * HD;
        op[lm]      = o0[r];
        op[16 + lm] = o1[r];
        if (lm == 0) {
            m_part[((size_t)seg * HEADS + head) * N_ROWS + row] = mrun;
            l_part[((size_t)seg * HEADS + head) * N_ROWS + row] = ls;
        }
    }
}

// ---------------------------------------------------------------------------
// Kernel 3: combine key-segments + o_proj + ReLU residual + W_out + log_softmax.
// ---------------------------------------------------------------------------
__global__ __launch_bounds__(128) void head_kernel(
    const float* __restrict__ h,
    const float* __restrict__ o_part, const float* __restrict__ m_part,
    const float* __restrict__ l_part,
    const float* __restrict__ opw, const float* __restrict__ opb,
    const float* __restrict__ Wout, const float* __restrict__ bout,
    float* __restrict__ out)
{
    __shared__ float os[HID];
    __shared__ float hs[HID];
    __shared__ float lgt[OUT_DIM];
    __shared__ float stat[2];
    const int row = blockIdx.x, t = threadIdx.x;
    const int hh = t >> 5;                        // head owning this dim

    // segment combine: O = sum_s e^(m_s-M) O_s / sum_s e^(m_s-M) l_s
    const float m0 = m_part[((size_t)0 * HEADS + hh) * N_ROWS + row];
    const float m1 = m_part[((size_t)1 * HEADS + hh) * N_ROWS + row];
    const float l0 = l_part[((size_t)0 * HEADS + hh) * N_ROWS + row];
    const float l1 = l_part[((size_t)1 * HEADS + hh) * N_ROWS + row];
    const float M  = fmaxf(m0, m1);
    const float w0 = __expf(m0 - M), w1 = __expf(m1 - M);
    const float num = w0 * o_part[((size_t)0 * N_ROWS + row) * HID + t]
                    + w1 * o_part[((size_t)1 * N_ROWS + row) * HID + t];
    os[t] = num * __builtin_amdgcn_rcpf(w0 * l0 + w1 * l1);
    __syncthreads();

    float a = opb[t];
    #pragma unroll 8
    for (int i = 0; i < HID; ++i) a += os[i] * opw[t * HID + i];
    hs[t] = h[(size_t)row * HID + t] + fmaxf(a, 0.0f);
    __syncthreads();

    if (t < OUT_DIM) {
        float g = bout[t];
        #pragma unroll 8
        for (int i = 0; i < HID; ++i) g += hs[i] * Wout[t * HID + i];
        lgt[t] = g;
    }
    __syncthreads();

    if (t == 0) {
        float m = -__builtin_inff();
        for (int j = 0; j < OUT_DIM; ++j) m = fmaxf(m, lgt[j]);
        float s = 0.0f;
        for (int j = 0; j < OUT_DIM; ++j) s += __expf(lgt[j] - m);
        stat[0] = m;  stat[1] = __logf(s);
    }
    __syncthreads();

    if (t < OUT_DIM) out[(size_t)row * OUT_DIM + t] = lgt[t] - stat[0] - stat[1];
}

// ---------------------------------------------------------------------------
extern "C" void kernel_launch(void* const* d_in, const int* in_sizes, int n_in,
                              void* d_out, int out_size, void* d_ws, size_t ws_size,
                              hipStream_t stream)
{
    const float* x    = (const float*)d_in[0];
    /* d_in[1] pos_enc unused (faithful to reference) */
    const float* sp   = (const float*)d_in[2];
    const float* W_in = (const float*)d_in[3];
    const float* b_in = (const float*)d_in[4];
    const float* ipw  = (const float*)d_in[5];
    const float* ipb  = (const float*)d_in[6];
    const float* opw  = (const float*)d_in[7];
    const float* opb  = (const float*)d_in[8];
    const float* Wout = (const float*)d_in[9];
    const float* bout = (const float*)d_in[10];
    const float* bsc  = (const float*)d_in[11];
    float* out = (float*)d_out;

    // workspace layout
    float*    h_ws = (float*)d_ws;                                    // N*HID
    float*    o_part = h_ws + (size_t)N_ROWS * HID;                   // SEG*N*HID
    float*    m_part = o_part + (size_t)SEG * N_ROWS * HID;           // SEG*HEADS*N
    float*    l_part = m_part + (size_t)SEG * HEADS * N_ROWS;         // SEG*HEADS*N
    _Float16* q_ws = (_Float16*)(l_part + (size_t)SEG * HEADS * N_ROWS);
    _Float16* k_ws = q_ws + (size_t)HEADS * N_ROWS * HD;
    _Float16* v_ws = k_ws + (size_t)HEADS * N_ROWS * HD;

    proj_qkv_kernel<<<N_ROWS, 128, 0, stream>>>(x, W_in, b_in, ipw, ipb,
                                                h_ws, q_ws, k_ws, v_ws);
    flash_attn_kernel<<<dim3(N_ROWS / QBLK, SEG), 512, 0, stream>>>(
        sp, bsc, q_ws, k_ws, v_ws, o_part, m_part, l_part);
    head_kernel<<<N_ROWS, 128, 0, stream>>>(h_ws, o_part, m_part, l_part,
                                            opw, opb, Wout, bout, out);
}
